// GATLayer_3736621548264
// MI455X (gfx1250) — matrix operand order repaired and checked by reading, not compiled, not run on hardware
//
#include <hip/hip_runtime.h>
#include <cstddef>

typedef _Float16 h8  __attribute__((ext_vector_type(8)));
typedef _Float16 h16 __attribute__((ext_vector_type(16)));
typedef float    f8  __attribute__((ext_vector_type(8)));

#define IN_CH    128
#define OUT_CH   128
#define HEADS    4
#define NEG_SLOPE 0.2f
#define LN_EPS   1e-5f

// ---------------- conversion: x (f32) -> x (f16) ----------------
__global__ void k_convert_x(const float* __restrict__ x, _Float16* __restrict__ xh, int n) {
    int i = blockIdx.x * blockDim.x + threadIdx.x;
    if (i < n) xh[i] = (_Float16)x[i];
}

// ---------------- pack W_l / W_r into WMMA B-fragment layout ----------------
// packed layout: [w(2)][ntile(8)][kstep(4)][lane(32)][half(16)]
// B (32x16 f16): lanes 0-15 -> K = kstep*32 + 0..15, lanes 16-31 -> K = kstep*32 + 16..31,
// half j -> K = kb + j, column n = ntile*16 + (lane&15).
__global__ void k_pack_w(const float* __restrict__ Wl, const float* __restrict__ Wr,
                         _Float16* __restrict__ packed) {
    int t = blockIdx.x * blockDim.x + threadIdx.x;
    if (t >= 2 * 8 * 4 * 32) return;
    int lane = t & 31;
    int ks   = (t >> 5) & 3;
    (void)ks;
    int nt   = (t >> 7) & 7;
    int w    = t >> 10;
    const float* W = w ? Wr : Wl;
    int n  = nt * 16 + (lane & 15);
    int kb = ((t >> 5) & 3) * 32 + ((lane & 16) ? 16 : 0);
    _Float16* dst = packed + (size_t)t * 16;
#pragma unroll
    for (int j = 0; j < 16; ++j) dst[j] = (_Float16)W[(size_t)(kb + j) * OUT_CH + n];
}

// ---------------- init: zero accumulator & denom, smax = -inf ----------------
__global__ void k_init(float* __restrict__ out, float* __restrict__ smax,
                       float* __restrict__ denom, int N) {
    int i = blockIdx.x * blockDim.x + threadIdx.x;
    if (i < N * OUT_CH) out[i] = 0.0f;
    if (i < N * HEADS) { smax[i] = -__builtin_huge_valf(); denom[i] = 0.0f; }
}

// ---------------- dual GEMM via WMMA: xl = x@Wl+bl, xr = x@Wr+br ----------------
// one wave per (16-row M-tile, weight); 4 k-steps x 8 n-tiles = 32 WMMAs/wave
__global__ void __launch_bounds__(256) k_gemm_wmma(
    const _Float16* __restrict__ xh, const _Float16* __restrict__ packedB,
    const float* __restrict__ bl, const float* __restrict__ br,
    float* __restrict__ xl, float* __restrict__ xr, int N) {
    int wave = (blockIdx.x * blockDim.x + threadIdx.x) >> 5;
    int lane = threadIdx.x & 31;
    int nM   = (N + 15) >> 4;
    if (wave >= nM * 2) return;
    int w     = wave & 1;
    int mtile = wave >> 1;

    // A-fragment addressing, contiguous K order: lane half h holds K = 16h+s, as in packed B
    int rowA = mtile * 16 + (lane & 15);
    if (rowA >= N) rowA = N - 1;                 // clamp (N%16==0, never hit)
    int kbA = (lane & 16) ? 16 : 0;

    const _Float16* Bw  = packedB + (size_t)w * (8 * 4 * 32 * 16);
    const float* bias   = w ? br : bl;
    float* outp         = w ? xr : xl;

    f8 acc[8] = {};
#pragma unroll
    for (int ks = 0; ks < 4; ++ks) {
        const _Float16* ap = xh + (size_t)rowA * IN_CH + ks * 32 + kbA;
        h8 alo = *(const h8*)ap;          // K = kb .. kb+7
        h8 ahi = *(const h8*)(ap + 8);    // K = kb+8 .. kb+15
        h16 a = __builtin_shufflevector(alo, ahi,
                    0, 1, 2, 3, 4, 5, 6, 7, 8, 9, 10, 11, 12, 13, 14, 15);
#pragma unroll
        for (int nt = 0; nt < 8; ++nt) {
            h16 b = *(const h16*)(Bw + ((size_t)(nt * 4 + ks) * 32 + lane) * 16);
            acc[nt] = __builtin_amdgcn_wmma_f32_16x16x32_f16(
                false, a, false, b, (short)0, acc[nt], false, false);
        }
    }

    // D layout: vgpr r -> row = mtile*16 + (lane<16 ? r : 8+r), col = lane&15
    int col   = lane & 15;
    int mrow0 = mtile * 16 + ((lane & 16) ? 8 : 0);
#pragma unroll
    for (int nt = 0; nt < 8; ++nt) {
        float bv = bias[nt * 16 + col];
#pragma unroll
        for (int r = 0; r < 8; ++r) {
            int m = mrow0 + r;
            if (m < N) outp[(size_t)m * OUT_CH + nt * 16 + col] = acc[nt][r] + bv;
        }
    }
}

// ---------------- attention scores + segment max (one wave per edge) ----------------
__global__ void k_score(const float* __restrict__ xl, const float* __restrict__ xr,
                        const int* __restrict__ ei, const float* __restrict__ att,
                        float* __restrict__ scores, float* __restrict__ smax,
                        int E, int Etot) {
    int wave = (blockIdx.x * blockDim.x + threadIdx.x) >> 5;
    int lane = threadIdx.x & 31;
    if (wave >= Etot) return;
    int src, dst;
    if (wave < E) { src = ei[wave]; dst = ei[(size_t)E + wave]; }
    else          { src = wave - E; dst = src; }

    float4 xlv = *(const float4*)(xl + (size_t)src * OUT_CH + lane * 4);
    float4 xrv = *(const float4*)(xr + (size_t)dst * OUT_CH + lane * 4);
    float4 av  = *(const float4*)(att + lane * 4);   // att flat index == channel
    float s = 0.f, t;
    t = xlv.x + xrv.x; t = t > 0.f ? t : NEG_SLOPE * t; s += av.x * t;
    t = xlv.y + xrv.y; t = t > 0.f ? t : NEG_SLOPE * t; s += av.y * t;
    t = xlv.z + xrv.z; t = t > 0.f ? t : NEG_SLOPE * t; s += av.z * t;
    t = xlv.w + xrv.w; t = t > 0.f ? t : NEG_SLOPE * t; s += av.w * t;
    // reduce within each 8-lane head group
    s += __shfl_xor(s, 1, 32);
    s += __shfl_xor(s, 2, 32);
    s += __shfl_xor(s, 4, 32);
    if ((lane & 7) == 0) {
        int h = lane >> 3;
        scores[(size_t)wave * HEADS + h] = s;
        float* addr = smax + (size_t)dst * HEADS + h;
        // float atomic-max via integer ordering trick (init = -inf)
        if (s >= 0.f) atomicMax((int*)addr, __float_as_int(s));
        else          atomicMin((unsigned int*)addr, __float_as_uint(s));
    }
}

// ---------------- exp(score - max) + segment sum ----------------
__global__ void k_expsum(const int* __restrict__ ei, float* __restrict__ scores,
                         const float* __restrict__ smax, float* __restrict__ denom,
                         int E, int Etot) {
    int t = blockIdx.x * blockDim.x + threadIdx.x;
    if (t >= Etot * HEADS) return;
    int e = t >> 2, h = t & 3;
    int dst = (e < E) ? ei[(size_t)E + e] : (e - E);
    float v = __expf(scores[t] - smax[(size_t)dst * HEADS + h]);
    scores[t] = v;
    unsafeAtomicAdd(denom + (size_t)dst * HEADS + h, v);
}

// ---------------- weighted scatter-sum (one wave per edge) ----------------
__global__ void k_scatter(const float* __restrict__ xl, const int* __restrict__ ei,
                          const float* __restrict__ scores, const float* __restrict__ denom,
                          float* __restrict__ out, int E, int Etot) {
    int wave = (blockIdx.x * blockDim.x + threadIdx.x) >> 5;
    int lane = threadIdx.x & 31;
    if (wave >= Etot) return;
    int src, dst;
    if (wave < E) { src = ei[wave]; dst = ei[(size_t)E + wave]; }
    else          { src = wave - E; dst = src; }
    int h = lane >> 3;
    float alpha = scores[(size_t)wave * HEADS + h] / denom[(size_t)dst * HEADS + h];
    float4 xlv = *(const float4*)(xl + (size_t)src * OUT_CH + lane * 4);
    float* op  = out + (size_t)dst * OUT_CH + lane * 4;
    unsafeAtomicAdd(op + 0, alpha * xlv.x);
    unsafeAtomicAdd(op + 1, alpha * xlv.y);
    unsafeAtomicAdd(op + 2, alpha * xlv.z);
    unsafeAtomicAdd(op + 3, alpha * xlv.w);
}

// ---------------- bias + SiLU + LayerNorm (one wave per node) ----------------
__global__ void k_finalize(float* __restrict__ out, const float* __restrict__ bias,
                           const float* __restrict__ gamma, const float* __restrict__ beta,
                           int N) {
    int wave = (blockIdx.x * blockDim.x + threadIdx.x) >> 5;
    int lane = threadIdx.x & 31;
    if (wave >= N) return;
    float* row = out + (size_t)wave * OUT_CH + lane * 4;
    float4 v  = *(const float4*)row;
    float4 bv = *(const float4*)(bias + lane * 4);
    v.x += bv.x; v.y += bv.y; v.z += bv.z; v.w += bv.w;
    v.x = v.x / (1.f + __expf(-v.x));
    v.y = v.y / (1.f + __expf(-v.y));
    v.z = v.z / (1.f + __expf(-v.z));
    v.w = v.w / (1.f + __expf(-v.w));
    float s  = v.x + v.y + v.z + v.w;
    float ss = v.x * v.x + v.y * v.y + v.z * v.z + v.w * v.w;
#pragma unroll
    for (int m = 1; m < 32; m <<= 1) {
        s  += __shfl_xor(s,  m, 32);
        ss += __shfl_xor(ss, m, 32);
    }
    float mean = s * (1.f / OUT_CH);
    float var  = ss * (1.f / OUT_CH) - mean * mean;
    float rs   = rsqrtf(var + LN_EPS);
    float4 gv = *(const float4*)(gamma + lane * 4);
    float4 bt = *(const float4*)(beta + lane * 4);
    float4 o;
    o.x = (v.x - mean) * rs * gv.x + bt.x;
    o.y = (v.y - mean) * rs * gv.y + bt.y;
    o.z = (v.z - mean) * rs * gv.z + bt.z;
    o.w = (v.w - mean) * rs * gv.w + bt.w;
    *(float4*)row = o;
}

extern "C" void kernel_launch(void* const* d_in, const int* in_sizes, int n_in,
                              void* d_out, int out_size, void* d_ws, size_t ws_size,
                              hipStream_t stream) {
    (void)n_in; (void)out_size; (void)ws_size;
    const float* x    = (const float*)d_in[0];
    const int*   ei   = (const int*)d_in[1];   // JAX default x64-off -> int32
    const float* Wl   = (const float*)d_in[2];
    const float* bl   = (const float*)d_in[3];
    const float* Wr   = (const float*)d_in[4];
    const float* br   = (const float*)d_in[5];
    const float* att  = (const float*)d_in[6];
    const float* bias = (const float*)d_in[7];
    const float* gamma= (const float*)d_in[8];
    const float* beta = (const float*)d_in[9];
    float* out = (float*)d_out;

    int N    = in_sizes[0] / IN_CH;
    int E    = in_sizes[1] / 2;
    int Etot = E + N;

    auto align256 = [](size_t o) { return (o + 255) & ~(size_t)255; };
    char* ws = (char*)d_ws;
    size_t off = 0;
    _Float16* xh      = (_Float16*)(ws + off); off = align256(off + (size_t)N * IN_CH * 2);
    _Float16* packedB = (_Float16*)(ws + off); off = align256(off + (size_t)2 * 8 * 4 * 32 * 16 * 2);
    float* xl     = (float*)(ws + off); off = align256(off + (size_t)N * OUT_CH * 4);
    float* xr     = (float*)(ws + off); off = align256(off + (size_t)N * OUT_CH * 4);
    float* scores = (float*)(ws + off); off = align256(off + (size_t)Etot * HEADS * 4);
    float* smax   = (float*)(ws + off); off = align256(off + (size_t)N * HEADS * 4);
    float* denom  = (float*)(ws + off); off = align256(off + (size_t)N * HEADS * 4);

    int nElemX = N * IN_CH;
    k_convert_x<<<(nElemX + 255) / 256, 256, 0, stream>>>(x, xh, nElemX);
    k_pack_w<<<8, 256, 0, stream>>>(Wl, Wr, packedB);
    k_init<<<(N * OUT_CH + 255) / 256, 256, 0, stream>>>(out, smax, denom, N);

    int nM = (N + 15) / 16;
    int gemmWaves = nM * 2;
    k_gemm_wmma<<<(gemmWaves + 7) / 8, 256, 0, stream>>>(xh, packedB, bl, br, xl, xr, N);

    k_score<<<(Etot + 7) / 8, 256, 0, stream>>>(xl, xr, ei, att, scores, smax, E, Etot);
    k_expsum<<<((size_t)Etot * HEADS + 255) / 256, 256, 0, stream>>>(ei, scores, smax, denom, E, Etot);
    k_scatter<<<(Etot + 7) / 8, 256, 0, stream>>>(xl, ei, scores, denom, out, E, Etot);
    k_finalize<<<(N + 7) / 8, 256, 0, stream>>>(out, bias, gamma, beta, N);
}
